// BasicShadowMamba_28647431864587
// MI455X (gfx1250) — compile-verified, matrix operand order checked by reading, not run on hardware
//
#include <hip/hip_runtime.h>
#include <hip/hip_bf16.h>
#include <math.h>

typedef __attribute__((ext_vector_type(16))) __bf16 v16bf;
typedef __attribute__((ext_vector_type(8)))  float  v8f;
typedef __attribute__((ext_vector_type(4)))  float  v4f;

static constexpr int Bn  = 4,  Cc = 48, Hh = 96, Wd = 96;
static constexpr int DIN = 96, Ns = 16, Rr = 3,  Kg = 4, HD = 192;
static constexpr int Ll  = Hh * Wd;        // 9216
static constexpr int HWp = Hh * Wd;        // 9216
static constexpr int Mpix = Bn * HWp;      // 36864
static constexpr int NCk = 96, CT = 96;    // scan chunks / chunk length

__device__ __forceinline__ float silu_f(float x)     { return x / (1.f + __expf(-x)); }
__device__ __forceinline__ float softplus_f(float x) { return (x > 20.f) ? x : log1pf(__expf(x)); }

// ---------------------------------------------------------------------------
// Weight pre-swizzle: fp32 row-major (Nout, Kin) -> bf16 WMMA B-tiles.
// Tile (nt,kt) = 512 halfs; lane L holds col n=nt*16+(L&15); half pair p holds
// K = kt*32 + (p&3)*2 + (L&16 ? 8:0) + (p&4 ? 16:0)  (+0,+1).  Zero-padded.
// ---------------------------------------------------------------------------
__global__ void pack_w_kernel(const float* __restrict__ W, __bf16* __restrict__ out,
                              int Nreal, int Kreal, int NT, int KT) {
  int t = blockIdx.x * blockDim.x + threadIdx.x;
  int total = NT * KT * 32;
  if (t >= total) return;
  int lane = t & 31, tile = t >> 5;
  int kt = tile % KT, nt = tile / KT;
  int n = nt * 16 + (lane & 15);
  int khalf = (lane & 16) ? 8 : 0;
  __bf16* o = out + (long)tile * 512 + lane * 16;
#pragma unroll
  for (int p = 0; p < 8; ++p) {
    int k = kt * 32 + ((p & 3) << 1) + khalf + ((p & 4) ? 16 : 0);
    float v0 = (n < Nreal && k     < Kreal) ? W[(long)n * Kreal + k]     : 0.f;
    float v1 = (n < Nreal && k + 1 < Kreal) ? W[(long)n * Kreal + k + 1] : 0.f;
    o[2 * p]     = (__bf16)v0;
    o[2 * p + 1] = (__bf16)v1;
  }
}

// A = -exp(A_logs)
__global__ void aprep_kernel(const float* __restrict__ alog, float* __restrict__ aneg, int n) {
  int i = blockIdx.x * blockDim.x + threadIdx.x;
  if (i < n) aneg[i] = -__expf(alog[i]);
}

// ---------------------------------------------------------------------------
// Generic bf16 WMMA GEMM:  out[m,n] = sum_k A[m,k] * Wrow[n,k]  (+bias)
// A: bf16 row-major (Mrows, KT*32).  Wp: pre-swizzled tiles.  fp32 out.
// SM: 0 = row-major (M,Nout); 1 = (B,Nout,96,96); 2 = accumulate into CHW.
// For SM 1/2 each lane's 8 results are 8 consecutive pixels of one channel,
// 32B-aligned -> vector b128 stores, no division anywhere.
// ---------------------------------------------------------------------------
template <int KT, int SM, bool BIAS>
__global__ void gemm_bf16_wmma(const __bf16* __restrict__ A,
                               const __bf16* __restrict__ Wp,
                               const float* __restrict__ bias,
                               float* __restrict__ out,
                               int Mrows, int Nout, int NT,
                               long aBatch, long oBatch) {
  const int lane = threadIdx.x & 31;
  const int wave = threadIdx.x >> 5;
  const int m0 = (blockIdx.x * (int)(blockDim.x >> 5) + wave) * 16;
  if (m0 >= Mrows) return;
  const __bf16* Ab = A + (long)blockIdx.z * aBatch;
  float* Ob = out + (long)blockIdx.z * oBatch;
  const int lda = KT * 32;
  const int row = m0 + (lane & 15);
  const int khalf = (lane & 16) ? 8 : 0;
  const int mhi = (lane & 16) ? 8 : 0;
  const int bb = m0 / HWp;          // constexpr divisor -> mul/shift
  const int pixbase = m0 % HWp;     // whole 16-row tile is inside one image

  union { v16bf v; unsigned int u[8]; } a[KT];
#pragma unroll
  for (int kt = 0; kt < KT; ++kt) {
#pragma unroll
    for (int p = 0; p < 8; ++p) {
      const int k = kt * 32 + ((p & 3) << 1) + khalf + ((p & 4) ? 16 : 0);
      a[kt].u[p] = *(const unsigned int*)(Ab + (long)row * lda + k);
    }
  }

  for (int nt = 0; nt < NT; ++nt) {
    if (nt + 1 < NT)  // stream next weight panel through L2 (global_prefetch_b8)
      __builtin_prefetch(Wp + ((long)(nt + 1) * KT) * 512 + lane * 16, 0, 0);
    const int ncol = nt * 16 + (lane & 15);
    v8f acc;
    const float bv = BIAS ? bias[ncol] : 0.f;
#pragma unroll
    for (int i = 0; i < 8; ++i) acc[i] = bv;
#pragma unroll
    for (int kt = 0; kt < KT; ++kt) {
      const v16bf w = *(const v16bf*)(Wp + ((long)(nt * KT + kt)) * 512 + lane * 16);
      acc = __builtin_amdgcn_wmma_f32_16x16x32_bf16(false, a[kt].v, false, w,
                                                    (short)0, acc, false, false);
    }
    if (SM == 0) {
#pragma unroll
      for (int r = 0; r < 8; ++r)
        Ob[(long)(m0 + mhi + r) * Nout + ncol] = acc[r];
    } else {
      float* cb = Ob + ((long)bb * Nout + ncol) * HWp + pixbase + mhi;
      v4f lo = {acc[0], acc[1], acc[2], acc[3]};
      v4f hi = {acc[4], acc[5], acc[6], acc[7]};
      if (SM == 2) {
        lo += *(const v4f*)cb;
        hi += *(const v4f*)(cb + 4);
      }
      *(v4f*)cb       = lo;
      *(v4f*)(cb + 4) = hi;
    }
  }
}

// LayerNorm over C=48 (read CHW-strided), write bf16 channel-last padded to 64.
__global__ void ln_to_bf16_kernel(const float* __restrict__ x,
                                  const float* __restrict__ w, const float* __restrict__ b,
                                  __bf16* __restrict__ out) {
  int pix = blockIdx.x * blockDim.x + threadIdx.x;
  if (pix >= Mpix) return;
  int bb = pix / HWp, p = pix % HWp;
  const float* xb = x + (long)bb * Cc * HWp + p;
  float s = 0.f, s2 = 0.f;
#pragma unroll 4
  for (int c = 0; c < Cc; ++c) { float v = xb[(long)c * HWp]; s += v; s2 += v * v; }
  float mu = s / Cc;
  float var = s2 / Cc - mu * mu;
  float inv = rsqrtf(var + 1e-5f);
  __bf16* o = out + (long)pix * 64;
#pragma unroll 4
  for (int c = 0; c < Cc; ++c)
    o[c] = (__bf16)((xb[(long)c * HWp] - mu) * inv * w[c] + b[c]);
#pragma unroll
  for (int c = Cc; c < 64; ++c) o[c] = (__bf16)0.f;
}

// Depthwise 3x3 SAME conv on 96x96, generic input strides, CHW out, opt SiLU.
template <int Ch, int SILU>
__global__ void dwconv3x3_kernel(const float* __restrict__ in,
                                 const float* __restrict__ wt, const float* __restrict__ bias,
                                 float* __restrict__ out,
                                 long inB, long inC, long inP) {
  int i = blockIdx.x * blockDim.x + threadIdx.x;
  if (i >= Bn * Ch * HWp) return;
  int w = i % Wd, h = (i / Wd) % Hh, c = (i / HWp) % Ch, b = i / (HWp * Ch);
  const float* wp = wt + c * 9;
  const float* ib = in + (long)b * inB + (long)c * inC;
  float s = bias[c];
#pragma unroll
  for (int dy = -1; dy <= 1; ++dy) {
    int yy = h + dy; if (yy < 0 || yy >= Hh) continue;
#pragma unroll
    for (int dx = -1; dx <= 1; ++dx) {
      int xx = w + dx; if (xx < 0 || xx >= Wd) continue;
      s += ib[(long)(yy * Wd + xx) * inP] * wp[(dy + 1) * 3 + (dx + 1)];
    }
  }
  if (SILU) s = silu_f(s);
  out[((long)(b * Ch + c) * Hh + h) * Wd + w] = s;
}

// CHW -> 4-direction cross-scan, bf16 (B,K,L,DIN) channel-last.
__global__ void cross_scan_kernel(const float* __restrict__ xm, __bf16* __restrict__ xs) {
  int i = blockIdx.x * blockDim.x + threadIdx.x;
  if (i >= Bn * DIN * HWp) return;
  int w = i % Wd, h = (i / Wd) % Hh, d = (i / HWp) % DIN, b = i / (HWp * DIN);
  float v = xm[i];
  int l0 = h * Wd + w;
  int l1 = w * Hh + h;
  long base = (long)b * Kg * Ll * DIN;
  xs[base + ((long)0 * Ll + l0) * DIN + d]            = (__bf16)v;
  xs[base + ((long)1 * Ll + l1) * DIN + d]            = (__bf16)v;
  xs[base + ((long)2 * Ll + (Ll - 1 - l0)) * DIN + d] = (__bf16)v;
  xs[base + ((long)3 * Ll + (Ll - 1 - l1)) * DIN + d] = (__bf16)v;
}

// delta = softplus(x_dbl[:, :R] @ dt_w^T + dt_b)
__global__ void delta_kernel(const float* __restrict__ xdbl, const float* __restrict__ dtw,
                             const float* __restrict__ dtb, float* __restrict__ delta) {
  int i = blockIdx.x * blockDim.x + threadIdx.x;
  if (i >= Bn * Kg * Ll * DIN) return;
  int d = i % DIN;
  long bkl = i / DIN;                 // (b*K + k)*L + l
  int k = (int)((bkl / Ll) & 3);
  const float* xr = xdbl + bkl * 48;
  const float* wr = dtw + ((long)k * DIN + d) * Rr;
  float s = dtb[k * DIN + d];
#pragma unroll
  for (int r = 0; r < Rr; ++r) s += xr[r] * wr[r];
  delta[i] = softplus_f(s);
}

// ---- chunked diagonal linear-recurrence scan ------------------------------
// block = 96 threads (one per d); blockIdx.x encodes (b,k,chunk)
__global__ void scan_pass1(const __bf16* __restrict__ xs, const float* __restrict__ delta,
                           const float* __restrict__ xdbl, const float* __restrict__ aneg,
                           float* __restrict__ aprodO, float* __restrict__ SO) {
  int d = threadIdx.x;
  int chunk = blockIdx.x % NCk;
  int k = (blockIdx.x / NCk) & 3;
  int b = blockIdx.x / (NCk * Kg);
  long seq = (long)b * Kg + k;
  long lbase = seq * Ll + (long)chunk * CT;
  const float*  dl = delta + lbase * DIN + d;
  const __bf16* us = xs    + lbase * DIN + d;
  const float*  bc = xdbl  + lbase * 48;
  float Av[Ns], ap[Ns], S[Ns];
#pragma unroll
  for (int n = 0; n < Ns; ++n) {
    Av[n] = aneg[((long)k * DIN + d) * Ns + n];
    ap[n] = 1.f; S[n] = 0.f;
  }
  for (int t = 0; t < CT; ++t) {
    __builtin_prefetch(dl + (long)(t + 8) * DIN, 0, 0);
    float dt_ = dl[(long)t * DIN];
    float u   = (float)us[(long)t * DIN];
    float du  = dt_ * u;
    const float* bv = bc + (long)t * 48 + Rr;
#pragma unroll
    for (int n = 0; n < Ns; ++n) {
      float a = __expf(dt_ * Av[n]);
      S[n]  = a * S[n] + du * bv[n];
      ap[n] *= a;
    }
  }
  long ob = (seq * DIN + d) * (long)(NCk * Ns) + (long)chunk * Ns;
#pragma unroll
  for (int n = 0; n < Ns; ++n) { aprodO[ob + n] = ap[n]; SO[ob + n] = S[n]; }
}

__global__ void scan_pass2(const float* __restrict__ aprod, const float* __restrict__ S,
                           float* __restrict__ hin) {
  int g = blockIdx.x * blockDim.x + threadIdx.x;   // (b*K+k)*DIN + d
  if (g >= Bn * Kg * DIN) return;
  long base = (long)g * (NCk * Ns);
  float h[Ns];
#pragma unroll
  for (int n = 0; n < Ns; ++n) h[n] = 0.f;
  for (int c = 0; c < NCk; ++c) {
    long o = base + (long)c * Ns;
#pragma unroll
    for (int n = 0; n < Ns; ++n) {
      hin[o + n] = h[n];
      h[n] = aprod[o + n] * h[n] + S[o + n];
    }
  }
}

__global__ void scan_pass3(const __bf16* __restrict__ xs, const float* __restrict__ delta,
                           const float* __restrict__ xdbl, const float* __restrict__ aneg,
                           const float* __restrict__ hin, const float* __restrict__ Ds,
                           float* __restrict__ ys) {
  int d = threadIdx.x;
  int chunk = blockIdx.x % NCk;
  int k = (blockIdx.x / NCk) & 3;
  int b = blockIdx.x / (NCk * Kg);
  long seq = (long)b * Kg + k;
  long lbase = seq * Ll + (long)chunk * CT;
  const float*  dl = delta + lbase * DIN + d;
  const __bf16* us = xs    + lbase * DIN + d;
  const float*  bc = xdbl  + lbase * 48;
  float Av[Ns], h[Ns];
  long hb = (seq * DIN + d) * (long)(NCk * Ns) + (long)chunk * Ns;
#pragma unroll
  for (int n = 0; n < Ns; ++n) {
    Av[n] = aneg[((long)k * DIN + d) * Ns + n];
    h[n]  = hin[hb + n];
  }
  float dsv = Ds[k * DIN + d];
  for (int t = 0; t < CT; ++t) {
    float dt_ = dl[(long)t * DIN];
    float u   = (float)us[(long)t * DIN];
    float du  = dt_ * u;
    const float* bv = bc + (long)t * 48 + Rr;
    const float* cv = bc + (long)t * 48 + Rr + Ns;
    float y = dsv * u;
#pragma unroll
    for (int n = 0; n < Ns; ++n) {
      float a = __expf(dt_ * Av[n]);
      h[n] = a * h[n] + du * bv[n];
      y += h[n] * cv[n];
    }
    ys[(lbase + t) * DIN + d] = y;   // aliases delta[] element-for-element: safe
  }
}

// cross-merge: ym[b,l,d] (HW-order, channel-last)
__global__ void merge_kernel(const float* __restrict__ ys, float* __restrict__ ym) {
  int i = blockIdx.x * blockDim.x + threadIdx.x;
  if (i >= Bn * Ll * DIN) return;
  int d = i % DIN;
  int l = (i / DIN) % Ll;
  int b = i / (DIN * Ll);
  int h = l / Wd, w = l % Wd;
  int lwh = w * Hh + h;
  long base = (long)b * Kg * Ll;
  float v = ys[(base + 0L * Ll + l) * DIN + d]
          + ys[(base + 1L * Ll + lwh) * DIN + d]
          + ys[(base + 2L * Ll + (Ll - 1 - l)) * DIN + d]
          + ys[(base + 3L * Ll + (Ll - 1 - lwh)) * DIN + d];
  ym[((long)b * Ll + l) * DIN + d] = v;
}

// out_norm LN over DIN + SiLU(z) gate -> bf16 (M,96)
__global__ void out_norm_gate_kernel(const float* __restrict__ ym,
                                     const float* __restrict__ w, const float* __restrict__ b,
                                     const float* __restrict__ xz, __bf16* __restrict__ out) {
  int row = blockIdx.x * blockDim.x + threadIdx.x;
  if (row >= Mpix) return;
  const float* y = ym + (long)row * DIN;
  float s = 0.f, s2 = 0.f;
#pragma unroll 4
  for (int c = 0; c < DIN; ++c) { float v = y[c]; s += v; s2 += v * v; }
  float mu = s / DIN;
  float inv = rsqrtf(s2 / DIN - mu * mu + 1e-5f);
  const float* zr = xz + (long)row * (2 * DIN) + DIN;
  __bf16* o = out + (long)row * DIN;
#pragma unroll 4
  for (int c = 0; c < DIN; ++c)
    o[c] = (__bf16)(((y[c] - mu) * inv * w[c] + b[c]) * silu_f(zr[c]));
}

// t3 = t1 * t2 from CHW (B,HD,96,96) -> bf16 channel-last (M, HD/2)
__global__ void gate_mul_kernel(const float* __restrict__ t, __bf16* __restrict__ out) {
  int i = blockIdx.x * blockDim.x + threadIdx.x;
  if (i >= Bn * (HD / 2) * HWp) return;
  int pix = i % HWp, c = (i / HWp) % (HD / 2), b = i / (HWp * (HD / 2));
  float t1 = t[((long)b * HD + c) * HWp + pix];
  float t2 = t[((long)b * HD + c + HD / 2) * HWp + pix];
  out[((long)b * HWp + pix) * (HD / 2) + c] = (__bf16)(t1 * t2);
}

// ---------------------------------------------------------------------------
extern "C" void kernel_launch(void* const* d_in, const int* in_sizes, int n_in,
                              void* d_out, int out_size, void* d_ws, size_t ws_size,
                              hipStream_t stream) {
  const float* x        = (const float*)d_in[0];
  const float* ln1_w    = (const float*)d_in[1];
  const float* ln1_b    = (const float*)d_in[2];
  const float* in_projw = (const float*)d_in[3];
  const float* conv_w   = (const float*)d_in[4];
  const float* conv_b   = (const float*)d_in[5];
  const float* x_projw  = (const float*)d_in[6];
  const float* dt_w     = (const float*)d_in[7];
  const float* dt_b     = (const float*)d_in[8];
  const float* A_logs   = (const float*)d_in[9];
  const float* Ds       = (const float*)d_in[10];
  const float* onorm_w  = (const float*)d_in[11];
  const float* onorm_b  = (const float*)d_in[12];
  const float* out_projw= (const float*)d_in[13];
  const float* ln2_w    = (const float*)d_in[14];
  const float* ln2_b    = (const float*)d_in[15];
  const float* lin1_w   = (const float*)d_in[16];
  const float* lin1_b   = (const float*)d_in[17];
  const float* dw_w     = (const float*)d_in[18];
  const float* dw_b     = (const float*)d_in[19];
  const float* lin2_w   = (const float*)d_in[20];
  const float* lin2_b   = (const float*)d_in[21];
  float* out = (float*)d_out;

  char* ws = (char*)d_ws;
  size_t off = 0;
  auto alloc = [&](size_t bytes) -> void* {
    void* p = ws + off;
    off = (off + bytes + 255) & ~(size_t)255;
    return p;
  };

  __bf16* pk_inproj  = (__bf16*)alloc(12 * 2 * 512 * 2);
  __bf16* pk_xproj   = (__bf16*)alloc(4 * 3 * 3 * 512 * 2);
  __bf16* pk_outproj = (__bf16*)alloc(3 * 3 * 512 * 2);
  __bf16* pk_lin1    = (__bf16*)alloc(12 * 2 * 512 * 2);
  __bf16* pk_lin2    = (__bf16*)alloc(3 * 3 * 512 * 2);
  float*  aneg       = (float*)alloc((size_t)Kg * DIN * Ns * 4);
  __bf16* xn_bf      = (__bf16*)alloc((size_t)Mpix * 64 * 2);            // reused: xn2
  float*  xz         = (float*)alloc((size_t)Mpix * 192 * 4);
  float*  xmconv     = (float*)alloc((size_t)Bn * DIN * HWp * 4);        // reused: ym
  __bf16* xs_bf      = (__bf16*)alloc((size_t)Bn * Kg * Ll * DIN * 2);   // reused: t (fp32 CHW)
  float*  xdbl       = (float*)alloc((size_t)Bn * Kg * Ll * 48 * 4);     // reused: tdw
  float*  delta      = (float*)alloc((size_t)Bn * Kg * Ll * DIN * 4);    // aliased: ys
  float*  aprod      = (float*)alloc((size_t)Bn * Kg * DIN * NCk * Ns * 4);
  float*  Sc         = (float*)alloc((size_t)Bn * Kg * DIN * NCk * Ns * 4);
  float*  hin        = (float*)alloc((size_t)Bn * Kg * DIN * NCk * Ns * 4);
  __bf16* ygate_bf   = (__bf16*)alloc((size_t)Mpix * DIN * 2);           // reused: t3
  (void)ws_size; (void)in_sizes; (void)n_in; (void)out_size;

  float*  ym   = xmconv;
  float*  t_f  = (float*)xs_bf;
  float*  tdw  = xdbl;
  float*  ys   = delta;
  __bf16* xn2  = xn_bf;
  __bf16* t3   = ygate_bf;

  const int TB = 256;
  auto blks = [](long n, int b) { return (unsigned)((n + b - 1) / b); };

  // --- weight packing + A precompute ---
  pack_w_kernel<<<blks(12 * 2 * 32, TB), TB, 0, stream>>>(in_projw, pk_inproj, 192, 48, 12, 2);
  for (int k = 0; k < Kg; ++k)
    pack_w_kernel<<<blks(3 * 3 * 32, TB), TB, 0, stream>>>(
        x_projw + (long)k * 35 * DIN, pk_xproj + (long)k * 3 * 3 * 512, 35, 96, 3, 3);
  pack_w_kernel<<<blks(3 * 3 * 32, TB), TB, 0, stream>>>(out_projw, pk_outproj, 48, 96, 3, 3);
  pack_w_kernel<<<blks(12 * 2 * 32, TB), TB, 0, stream>>>(lin1_w, pk_lin1, 192, 48, 12, 2);
  pack_w_kernel<<<blks(3 * 3 * 32, TB), TB, 0, stream>>>(lin2_w, pk_lin2, 48, 96, 3, 3);
  aprep_kernel<<<blks(Kg * DIN * Ns, TB), TB, 0, stream>>>(A_logs, aneg, Kg * DIN * Ns);

  // --- branch 1: LN1 -> in_proj (WMMA) -> dwconv+SiLU -> cross-scan ---
  ln_to_bf16_kernel<<<blks(Mpix, TB), TB, 0, stream>>>(x, ln1_w, ln1_b, xn_bf);
  gemm_bf16_wmma<2, 0, false><<<dim3(Mpix / 128, 1, 1), TB, 0, stream>>>(
      xn_bf, pk_inproj, nullptr, xz, Mpix, 192, 12, 0, 0);
  dwconv3x3_kernel<96, 1><<<blks((long)Bn * DIN * HWp, TB), TB, 0, stream>>>(
      xz, conv_w, conv_b, xmconv, (long)HWp * 192, 1L, 192L);
  cross_scan_kernel<<<blks((long)Bn * DIN * HWp, TB), TB, 0, stream>>>(xmconv, xs_bf);

  // --- x_proj (WMMA, per direction, batched over B) + delta ---
  for (int k = 0; k < Kg; ++k)
    gemm_bf16_wmma<3, 0, false><<<dim3(Ll / 128, 1, Bn), TB, 0, stream>>>(
        xs_bf + (long)k * Ll * DIN, pk_xproj + (long)k * 3 * 3 * 512, nullptr,
        xdbl + (long)k * Ll * 48, Ll, 48, 3,
        (long)Kg * Ll * DIN, (long)Kg * Ll * 48);
  delta_kernel<<<blks((long)Bn * Kg * Ll * DIN, TB), TB, 0, stream>>>(xdbl, dt_w, dt_b, delta);

  // --- chunked parallel selective scan ---
  scan_pass1<<<Bn * Kg * NCk, DIN, 0, stream>>>(xs_bf, delta, xdbl, aneg, aprod, Sc);
  scan_pass2<<<blks(Bn * Kg * DIN, DIN), DIN, 0, stream>>>(aprod, Sc, hin);
  scan_pass3<<<Bn * Kg * NCk, DIN, 0, stream>>>(xs_bf, delta, xdbl, aneg, hin, Ds, ys);

  // --- merge + out_norm*SiLU(z) -> out_proj (WMMA, accumulate residual) ---
  merge_kernel<<<blks((long)Bn * Ll * DIN, TB), TB, 0, stream>>>(ys, ym);
  out_norm_gate_kernel<<<blks(Mpix, TB), TB, 0, stream>>>(ym, onorm_w, onorm_b, xz, ygate_bf);
  hipMemcpyAsync(out, x, (size_t)Mpix * Cc * 4, hipMemcpyDeviceToDevice, stream);
  gemm_bf16_wmma<3, 2, false><<<dim3(Mpix / 128, 1, 1), TB, 0, stream>>>(
      ygate_bf, pk_outproj, nullptr, out, Mpix, Cc, 3, 0, 0);          // out = x1

  // --- branch 2: LN2 -> lin1 (WMMA, CHW) -> dwconv -> t1*t2 -> lin2 (WMMA, accum) ---
  ln_to_bf16_kernel<<<blks(Mpix, TB), TB, 0, stream>>>(out, ln2_w, ln2_b, xn2);
  gemm_bf16_wmma<2, 1, true><<<dim3(Mpix / 128, 1, 1), TB, 0, stream>>>(
      xn2, pk_lin1, lin1_b, t_f, Mpix, HD, 12, 0, 0);
  dwconv3x3_kernel<192, 0><<<blks((long)Bn * HD * HWp, TB), TB, 0, stream>>>(
      t_f, dw_w, dw_b, tdw, (long)HD * HWp, (long)HWp, 1L);
  gate_mul_kernel<<<blks((long)Bn * (HD / 2) * HWp, TB), TB, 0, stream>>>(tdw, t3);
  gemm_bf16_wmma<3, 2, true><<<dim3(Mpix / 128, 1, 1), TB, 0, stream>>>(
      t3, pk_lin2, lin2_b, out, Mpix, Cc, 3, 0, 0);                    // out = x1 + m
}
